// TargetExtractionBlock_18528488915006
// MI455X (gfx1250) — compile-verified, matrix operand order checked by reading, not run on hardware
//
#include <hip/hip_runtime.h>
#include <hip/hip_bf16.h>

// ---------------------------------------------------------------------------
// Problem constants (from reference): B=64, S=512, D=768, H=768, C=5
// ---------------------------------------------------------------------------
#define BB 64
#define SS 512
#define DD 768
#define HH 768
#define CC 5
#define MTOT (BB * SS)          // 32768 fused token rows

#define USE_TDM 1               // tensor_load_to_lds for W2 staging (6-arg form)

// WMMA / TDM vector types (probe-confirmed signatures, CDNA5 / gfx1250)
typedef __attribute__((ext_vector_type(16))) __bf16       v16bf;
typedef __attribute__((ext_vector_type(8)))  __bf16       bf16x8;
typedef __attribute__((ext_vector_type(8)))  float        v8f;
typedef __attribute__((ext_vector_type(4)))  unsigned int v4u;
typedef __attribute__((ext_vector_type(8)))  int          v8i;
typedef __attribute__((ext_vector_type(4)))  int          v4i;

__device__ __forceinline__ __bf16 f2bf(float x) { return (__bf16)x; }

// ---------------------------------------------------------------------------
// Kernel 1: fused  relu(X@W1+b1) @ W2 + b2 (+ boundary energies) -> pot
//
// 256 blocks x 256 threads (8 wave32s). Block M-tile = 128 rows, wave
// M-tile = 16 rows. N (=768) swept as 24 passes of TWO 16-wide tiles: the
// 32-column W1 panel is staged transposed (f32 -> bf16) in LDS and shared by
// 8 waves x 24 K-steps; one A fragment (global f32 -> bf16, matching the ISA
// 16-bit A layout) feeds two v_wmma_f32_16x16x32_bf16 issues. GEMM2 (C=5)
// is fused on the ReLU'd accumulators so h never touches memory. W2 arrives
// in LDS via the Tensor Data Mover (1-D descriptor) + s_wait_tensorcnt.
// ---------------------------------------------------------------------------
#define KPAD (DD + 8)           // bf16 elems per sBT row: 1552B pitch keeps
                                // row-parallel ds_load_b128 on distinct banks

__global__ void __launch_bounds__(256)
fused_mlp_pot_kernel(const float* __restrict__ hs,   // [MTOT, DD]
                     const float* __restrict__ W1,   // [DD, HH]
                     const float* __restrict__ b1,   // [HH]
                     const float* __restrict__ W2,   // [HH, CC]
                     const float* __restrict__ b2,   // [CC]
                     const float* __restrict__ lb,   // [CC]
                     const float* __restrict__ rb,   // [CC]
                     float* __restrict__ pot_out)    // [MTOT, CC]
{
    __shared__ __bf16 sBT[32 * KPAD];   // W1 panel, transposed [n][k]: 49.7 KB
    __shared__ float  sW2f[HH * CC];    // flat W2 copy (TDM dest):     15.4 KB

    const int tid  = threadIdx.x;
    const int wave = tid >> 5;
    const int lane = tid & 31;
    const int half = lane >> 4;         // 0: lanes 0-15, 1: lanes 16-31
    const int lr   = lane & 15;

    // ---- stage W2 into LDS ---------------------------------------------------
#if USE_TDM
    if (tid < 32) {                      // one wave issues one TDM op
        // D# group 0: count=1, lds_addr, global_addr, type=2 ("image")
        const unsigned long long ga = (unsigned long long)(uintptr_t)W2;
        const unsigned int ldsoff   = (unsigned int)(uintptr_t)(void*)sW2f; // addr[31:0] = LDS offset
        v4u g0;
        g0[0] = 1u;                                    // count=1, is_restore=0
        g0[1] = ldsoff;                                // lds_addr
        g0[2] = (unsigned int)ga;                      // global_addr[31:0]
        g0[3] = (unsigned int)((ga >> 32) & 0x01FFFFFFu) | 0x80000000u; // [120:96] | type=2
        // D# group 1: data_size=4B, 1-D tensor: dim0=tile0=3840, stride0=3840
        v8i g1;
        g1[0] = 0x00020000;                            // workgroup_mask=0, data_size=2 (4B)
        g1[1] = (int)((HH * CC & 0xFFFF) << 16);       // tensor_dim0[15:0] @ bits 63:48
        g1[2] = (int)(((HH * CC) >> 16) | (1 << 16));  // tensor_dim0[31:16], tensor_dim1=1
        g1[3] = (int)((HH * CC) << 16);                // tile_dim0=3840 @ bits 127:112
        g1[4] = 0;                                     // tile_dim1=0 (unused), tile_dim2=0
        g1[5] = HH * CC;                               // tensor_dim0_stride[31:0]
        g1[6] = 0;
        g1[7] = 0;
        v4i gz  = {0, 0, 0, 0};
        v8i gz8 = {0, 0, 0, 0, 0, 0, 0, 0};
        __builtin_amdgcn_tensor_load_to_lds(g0, g1, gz, gz, gz8, 0);
        __builtin_amdgcn_s_wait_tensorcnt(0);
    }
#else
    for (int i = tid; i < HH * CC; i += 256) sW2f[i] = W2[i];
#endif

    const int m0 = blockIdx.x * 128 + wave * 16;     // this wave's row tile
    const int mA = m0 + lr;                          // row held by this lane (A)
    const float* __restrict__ arow = hs + (size_t)mA * DD;

    // fused GEMM2 partials: 8 rows x 5 classes over this lane's column subset
    float potacc[8][CC];
#pragma unroll
    for (int r = 0; r < 8; ++r)
#pragma unroll
        for (int c = 0; c < CC; ++c) potacc[r][c] = 0.0f;

    for (int ntp = 0; ntp < HH / 32; ++ntp) {        // 24 passes of 2 N-tiles
        const int n0 = ntp * 32;

        // ---- stage W1[:, n0:n0+32] transposed as bf16 ------------------------
        __syncthreads();                 // previous pass's readers done
        for (int i = tid; i < DD * 32; i += 256) {
            const int n = i & 31, k = i >> 5;        // coalesced over n
            sBT[n * KPAD + k] = f2bf(W1[(size_t)k * HH + n0 + n]);
        }
        __syncthreads();

        // ---- K loop: one A fragment feeds two WMMAs --------------------------
        v8f acc0 = {0.f, 0.f, 0.f, 0.f, 0.f, 0.f, 0.f, 0.f};
        v8f acc1 = {0.f, 0.f, 0.f, 0.f, 0.f, 0.f, 0.f, 0.f};
        const __bf16* __restrict__ brow0 = &sBT[lr * KPAD];         // tile n0+lr
        const __bf16* __restrict__ brow1 = &sBT[(16 + lr) * KPAD];  // tile n0+16+lr

        for (int kt = 0; kt < DD / 32; ++kt) {
            const int k0 = kt * 32;
            const int ka = k0 + half * 8;            // ISA 16-bit A layout:
                                                     // half0: K{0..7,16..23}
                                                     // half1: K{8..15,24..31}
            __builtin_prefetch(arow + k0 + 32, 0, 1); // global_prefetch_b8

            // A fragment: 16 f32 -> 16 bf16 (aligned float4 loads)
            const float4 a0 = *(const float4*)(arow + ka);
            const float4 a1 = *(const float4*)(arow + ka + 4);
            const float4 a2 = *(const float4*)(arow + ka + 16);
            const float4 a3 = *(const float4*)(arow + ka + 20);
            v16bf af;
            af[0]  = f2bf(a0.x); af[1]  = f2bf(a0.y); af[2]  = f2bf(a0.z); af[3]  = f2bf(a0.w);
            af[4]  = f2bf(a1.x); af[5]  = f2bf(a1.y); af[6]  = f2bf(a1.z); af[7]  = f2bf(a1.w);
            af[8]  = f2bf(a2.x); af[9]  = f2bf(a2.y); af[10] = f2bf(a2.z); af[11] = f2bf(a2.w);
            af[12] = f2bf(a3.x); af[13] = f2bf(a3.y); af[14] = f2bf(a3.z); af[15] = f2bf(a3.w);

            // B fragments: aligned 16B LDS loads (ds_load_b128)
            const bf16x8 b0lo = *(const bf16x8*)(brow0 + ka);
            const bf16x8 b0hi = *(const bf16x8*)(brow0 + ka + 16);
            const bf16x8 b1lo = *(const bf16x8*)(brow1 + ka);
            const bf16x8 b1hi = *(const bf16x8*)(brow1 + ka + 16);
            v16bf bf0, bf1;
#pragma unroll
            for (int j = 0; j < 8; ++j) {
                bf0[j] = b0lo[j]; bf0[8 + j] = b0hi[j];
                bf1[j] = b1lo[j]; bf1[8 + j] = b1hi[j];
            }

            acc0 = __builtin_amdgcn_wmma_f32_16x16x32_bf16(
                false, af, false, bf0, (short)0, acc0, false, false);
            acc1 = __builtin_amdgcn_wmma_f32_16x16x32_bf16(
                false, af, false, bf1, (short)0, acc1, false, false);
        }

        // ---- fused epilogue: +b1, ReLU, rank-32 update into pot --------------
        const int   ng0 = n0 + lr;                   // columns this lane owns
        const int   ng1 = n0 + 16 + lr;
        const float bi0 = b1[ng0];
        const float bi1 = b1[ng1];
        float w20[CC], w21[CC];
#pragma unroll
        for (int c = 0; c < CC; ++c) {
            w20[c] = sW2f[ng0 * CC + c];
            w21[c] = sW2f[ng1 * CC + c];
        }
#pragma unroll
        for (int r = 0; r < 8; ++r) {
            float h0 = acc0[r] + bi0; h0 = h0 > 0.0f ? h0 : 0.0f;
            float h1 = acc1[r] + bi1; h1 = h1 > 0.0f ? h1 : 0.0f;
#pragma unroll
            for (int c = 0; c < CC; ++c)
                potacc[r][c] += h0 * w20[c] + h1 * w21[c];
        }
    }

    // ---- cross-lane sum over the 16 columns held inside each half ------------
#pragma unroll
    for (int r = 0; r < 8; ++r)
#pragma unroll
        for (int c = 0; c < CC; ++c) {
            float v = potacc[r][c];
            v += __shfl_xor(v, 1, 32);
            v += __shfl_xor(v, 2, 32);
            v += __shfl_xor(v, 4, 32);
            v += __shfl_xor(v, 8, 32);
            potacc[r][c] = v;
        }

    if (lr == 0) {
        // lane 0 -> rows m0..m0+7, lane 16 -> rows m0+8..m0+15  (WMMA D layout)
#pragma unroll
        for (int r = 0; r < 8; ++r) {
            const int mm = m0 + half * 8 + r;
            const int s  = mm & (SS - 1);            // timestep within sequence
            float* o = pot_out + (size_t)mm * CC;
#pragma unroll
            for (int c = 0; c < CC; ++c) {
                float v = potacc[r][c] + b2[c];
                if (s == 0)      v += lb[c];
                if (s == SS - 1) v += rb[c];
                o[c] = v;
            }
        }
    }
}

// ---------------------------------------------------------------------------
// Kernel 2: seq_length = sum(mask) per batch; copy chain_kernel to output
// ---------------------------------------------------------------------------
__global__ void tail_kernel(const unsigned char* __restrict__ mask,  // [BB,SS]
                            const float* __restrict__ chain,         // [CC,CC]
                            float* __restrict__ out_seq,             // [BB]
                            float* __restrict__ out_chain)           // [CC*CC]
{
    const int t = threadIdx.x;
    if (t < BB) {
        const unsigned char* mr = mask + (size_t)t * SS;
        int ssum = 0;
        for (int s = 0; s < SS; ++s) ssum += mr[s];
        out_seq[t] = (float)ssum;
    }
    if (t >= 64 && t < 64 + CC * CC) out_chain[t - 64] = chain[t - 64];
}

// ---------------------------------------------------------------------------
// Kernel 3: Viterbi decode. 1 thread per batch; trans in registers,
// uint8 backpointers in workspace, backtrace writes tags as floats.
// ---------------------------------------------------------------------------
__global__ void viterbi_kernel(const float* __restrict__ pot,     // [BB,SS,CC]
                               const float* __restrict__ trans,   // [CC,CC]
                               float* __restrict__ decoded,       // [BB,SS]
                               unsigned char* __restrict__ bp)    // [BB,SS-1,CC]
{
    const int b = blockIdx.x * blockDim.x + threadIdx.x;
    if (b >= BB) return;

    float tr[CC * CC];
#pragma unroll
    for (int i = 0; i < CC * CC; ++i) tr[i] = trans[i];

    const float* p = pot + (size_t)b * SS * CC;
    unsigned char* bpb = bp + (size_t)b * (SS - 1) * CC;

    float alpha[CC];
#pragma unroll
    for (int c = 0; c < CC; ++c) alpha[c] = p[c];

    for (int s = 1; s < SS; ++s) {
        float na[CC];
#pragma unroll
        for (int c = 0; c < CC; ++c) {
            float best = alpha[0] + tr[0 * CC + c];
            int   bi   = 0;
#pragma unroll
            for (int cp = 1; cp < CC; ++cp) {
                const float sc = alpha[cp] + tr[cp * CC + c];
                if (sc > best) { best = sc; bi = cp; }   // argmax: first wins ties
            }
            na[c] = best + p[s * CC + c];
            bpb[(s - 1) * CC + c] = (unsigned char)bi;
        }
#pragma unroll
        for (int c = 0; c < CC; ++c) alpha[c] = na[c];
    }

    int   last = 0;
    float bv   = alpha[0];
#pragma unroll
    for (int c = 1; c < CC; ++c)
        if (alpha[c] > bv) { bv = alpha[c]; last = c; }

    decoded[(size_t)b * SS + (SS - 1)] = (float)last;
    int tag = last;
    for (int s = SS - 2; s >= 0; --s) {
        tag = bpb[s * CC + tag];
        decoded[(size_t)b * SS + s] = (float)tag;
    }
}

// ---------------------------------------------------------------------------
// d_out layout (floats, reference tuple order):
//   [0      , 32768 )  decoded   (int tags stored as floats)
//   [32768  , 196608)  pot       [B,S,C]
//   [196608 , 196672)  seq_length
//   [196672 , 196697)  chain_kernel
// ---------------------------------------------------------------------------
extern "C" void kernel_launch(void* const* d_in, const int* in_sizes, int n_in,
                              void* d_out, int out_size, void* d_ws, size_t ws_size,
                              hipStream_t stream) {
    const float*         hs    = (const float*)d_in[0];
    const unsigned char* mask  = (const unsigned char*)d_in[1];
    const float*         W1    = (const float*)d_in[2];
    const float*         b1    = (const float*)d_in[3];
    const float*         W2    = (const float*)d_in[4];
    const float*         b2    = (const float*)d_in[5];
    const float*         chain = (const float*)d_in[6];
    const float*         lb    = (const float*)d_in[7];
    const float*         rbnd  = (const float*)d_in[8];

    float* out       = (float*)d_out;
    float* out_dec   = out;
    float* out_pot   = out + MTOT;
    float* out_seq   = out + MTOT + MTOT * CC;
    float* out_chain = out_seq + BB;

    unsigned char* bp_ws = (unsigned char*)d_ws;   // BB*(SS-1)*CC = 163,520 B

    // GEMM1 (bf16 WMMA, f32 acc) fused with ReLU + GEMM2 + boundaries
    fused_mlp_pot_kernel<<<MTOT / 128, 256, 0, stream>>>(
        hs, W1, b1, W2, b2, lb, rbnd, out_pot);

    // seq_length + chain_kernel passthrough
    tail_kernel<<<1, 128, 0, stream>>>(mask, chain, out_seq, out_chain);

    // Viterbi decode (reads pot just written into d_out; same-stream ordering)
    viterbi_kernel<<<2, 32, 0, stream>>>(out_pot, chain, out_dec, bp_ws);
}
